// MultiHeadAttention_30391188586935
// MI455X (gfx1250) — matrix operand order repaired and checked by reading, not compiled, not run on hardware
//
#include <hip/hip_runtime.h>
#include <hip/hip_bf16.h>
#include <cstdint>
#include <cstddef>

// ---------------------------------------------------------------------------
// MHA for MI455X (gfx1250, wave32, WMMA).
// BF16 operands / FP32 accumulate via v_wmma_f32_16x16x32_bf16.
// GEMMs: 128x128 block tiles, 256 thr (8 waves x 32x64 strips), double-
// buffered LDS (1 barrier/K-step), global prefetch one slab ahead.
// All LDS fragment reads are 16B-contiguous -> ds_load_b128.
// ---------------------------------------------------------------------------

typedef __attribute__((ext_vector_type(16))) __bf16 v16bf;
typedef __attribute__((ext_vector_type(8)))  float  v8f;

#define B_   4
#define S_   2048
#define D_   1024
#define H_   16
#define HD_  64
#define N3_  3072
#define SCALE_ 0.125f   // 64^-0.5

#define WMMA_BF16(a, b, c) \
  __builtin_amdgcn_wmma_f32_16x16x32_bf16(false, (a), false, (b), (short)0, (c), false, false)

// ---------------- fragment loaders (ISA 7.12.2 layouts, wave32) -------------
// A (16x32 bf16): lane holds row (lane%16); element e -> K =
//   ((e/2)>=4 ? 16:0) + (lane/16)*8 + ((e/2)%4)*2 + (e%2)
// => two contiguous 8-element (16B) runs per lane.
__device__ __forceinline__ v16bf load_A_frag(const __bf16* base, int stride,
                                             int row, int kbase, int hh) {
  v16bf a;
#pragma unroll
  for (int e = 0; e < 16; ++e) {
    int v = e >> 1, p = e & 1;
    int k = ((v >= 4) ? 16 : 0) + hh * 8 + (v & 3) * 2 + p;
    a[e] = base[row * stride + kbase + k];
  }
  return a;
}

// B (32x16 bf16) from N-major storage M[n][k] (we need B = M^T):
//   element e -> M[col][kbase + (e>=8 ? 16:0) + (lane/16)*8 + (e%8)]  (two 16B runs)
__device__ __forceinline__ v16bf load_Bt_frag(const __bf16* base, int stride,
                                              int kbase, int col, int hh) {
  v16bf b;
  const __bf16* p = base + col * stride + kbase + hh * 8;
#pragma unroll
  for (int e = 0; e < 16; ++e) b[e] = p[(e >= 8 ? 16 : 0) + (e & 7)];
  return b;
}

__device__ __forceinline__ float redmax16(float v) {
  v = fmaxf(v, __shfl_xor(v, 1, 32));
  v = fmaxf(v, __shfl_xor(v, 2, 32));
  v = fmaxf(v, __shfl_xor(v, 4, 32));
  v = fmaxf(v, __shfl_xor(v, 8, 32));
  return v;
}
__device__ __forceinline__ float redsum16(float v) {
  v += __shfl_xor(v, 1, 32);
  v += __shfl_xor(v, 2, 32);
  v += __shfl_xor(v, 4, 32);
  v += __shfl_xor(v, 8, 32);
  return v;
}

// ---------------------------- convert kernels -------------------------------
__global__ void f32_to_bf16_kernel(const float* __restrict__ in,
                                   __bf16* __restrict__ out, int n) {
  int i = blockIdx.x * blockDim.x + threadIdx.x;
  if (i < n) out[i] = (__bf16)in[i];
}

// out[n][k] = (bf16) in[k][n]   for a KxN fp32 matrix (tiled transpose)
__global__ __launch_bounds__(256)
void f32_to_bf16_T_kernel(const float* __restrict__ in, __bf16* __restrict__ out,
                          int K, int N) {
  __shared__ float tile[32][33];
  const int n0 = blockIdx.x * 32;
  const int k0 = blockIdx.y * 32;
  const int tx = threadIdx.x & 31;
  const int ty = threadIdx.x >> 5;          // 0..7
#pragma unroll
  for (int i = ty; i < 32; i += 8)
    tile[i][tx] = in[(size_t)(k0 + i) * N + n0 + tx];
  __syncthreads();
#pragma unroll
  for (int i = ty; i < 32; i += 8)
    out[(size_t)(n0 + i) * K + k0 + tx] = (__bf16)tile[tx][i];
}

// ------------------------------ QKV GEMM ------------------------------------
// C[8192,3072] = Xb[8192,1024] @ Wt^T + bias ; Wt is [3072][1024] (N-major).
// Block: 256 thr (8 waves), tile 128x128, K-step 32, double-buffered LDS.
// Q,K stored [B,H,S,HD]; V stored transposed [B,H,HD,S].
__global__ __launch_bounds__(256)
void qkv_gemm_kernel(const __bf16* __restrict__ X, const __bf16* __restrict__ Wt,
                     const float* __restrict__ bias,
                     __bf16* __restrict__ Qb, __bf16* __restrict__ Kb,
                     __bf16* __restrict__ Vt) {
  const int nBase = blockIdx.x * 128;
  const int mBase = blockIdx.y * 128;
  const int tid  = threadIdx.x;
  const int wv   = tid >> 5, lane = tid & 31;
  const int m16  = lane & 15, hh = lane >> 4;
  const int wrow = (wv & 3) * 32;     // wave's row-strip base (0..96)
  const int wcol = (wv >> 2) * 64;    // wave's col-half base (0/64)

  __shared__ __align__(16) __bf16 As[2][128][40];  // 2 x 128x32, stride 40
  __shared__ __align__(16) __bf16 Bs[2][128][40];

  const int ar = tid >> 1, acb = (tid & 1) * 16;   // 256 thr cover 128x32

  const __bf16* Xrow = X  + (size_t)(mBase + ar) * D_ + acb;
  const __bf16* Wrow = Wt + (size_t)(nBase + ar) * D_ + acb;

  // preload slab 0
  uint4 xr0, xr1, wr0, wr1;
  { const uint4* s = (const uint4*)(Xrow); xr0 = s[0]; xr1 = s[1]; }
  { const uint4* s = (const uint4*)(Wrow); wr0 = s[0]; wr1 = s[1]; }
  { uint4* d = (uint4*)&As[0][ar][acb]; d[0] = xr0; d[1] = xr1; }
  { uint4* d = (uint4*)&Bs[0][ar][acb]; d[0] = wr0; d[1] = wr1; }

  v8f c[2][4] = {};
  int cur = 0;

  for (int k0 = 0; k0 < D_; k0 += 32) {
    __syncthreads();
    const bool hasNext = (k0 + 32) < D_;
    if (hasNext) {
      { const uint4* s = (const uint4*)(Xrow + k0 + 32); xr0 = s[0]; xr1 = s[1]; }
      { const uint4* s = (const uint4*)(Wrow + k0 + 32); wr0 = s[0]; wr1 = s[1]; }
      if (k0 + 64 < D_) {                       // pull the slab after that to L2
        __builtin_prefetch(Xrow + k0 + 64, 0, 0);
        __builtin_prefetch(Wrow + k0 + 64, 0, 0);
      }
    }

    v16bf a0 = load_A_frag(&As[cur][0][0], 40, wrow + m16, 0, hh);
    v16bf a1 = load_A_frag(&As[cur][0][0], 40, wrow + 16 + m16, 0, hh);
#pragma unroll
    for (int f = 0; f < 4; ++f) {
      v16bf b = load_Bt_frag(&Bs[cur][0][0], 40, 0, wcol + f * 16 + m16, hh);
      c[0][f] = WMMA_BF16(a0, b, c[0][f]);
      c[1][f] = WMMA_BF16(a1, b, c[1][f]);
    }

    if (hasNext) {
      { uint4* d = (uint4*)&As[cur ^ 1][ar][acb]; d[0] = xr0; d[1] = xr1; }
      { uint4* d = (uint4*)&Bs[cur ^ 1][ar][acb]; d[0] = wr0; d[1] = wr1; }
      cur ^= 1;
    }
  }

  // Epilogue: bias add + scatter into Q/K [B,H,S,HD] and V^T [B,H,HD,S].
  const int cbase = nBase + wcol;                 // wave-uniform
  const int t = cbase >> 10;                      // 0=q, 1=k, 2=v (wave-uniform)
#pragma unroll
  for (int f = 0; f < 4; ++f) {
    int cc  = cbase + f * 16 + m16;
    float bv = bias[cc];
    int rem = cc & 1023;
    int h   = rem >> 6;
    int hd  = rem & 63;
    if (t == 2) {
#pragma unroll
      for (int r2 = 0; r2 < 2; ++r2)
#pragma unroll
        for (int i = 0; i < 8; ++i) {
          int r  = mBase + wrow + r2 * 16 + i + 8 * hh;
          int bb = r >> 11;
          int ss = r & (S_ - 1);
          Vt[((size_t)(bb * H_ + h) * HD_ + hd) * S_ + ss] = (__bf16)(c[r2][f][i] + bv);
        }
    } else {
      __bf16* dst = (t == 0) ? Qb : Kb;
#pragma unroll
      for (int r2 = 0; r2 < 2; ++r2)
#pragma unroll
        for (int i = 0; i < 8; ++i) {
          int r  = mBase + wrow + r2 * 16 + i + 8 * hh;
          int bb = r >> 11;
          int ss = r & (S_ - 1);
          dst[(((size_t)(bb * H_ + h) * S_ + ss) << 6) + hd] = (__bf16)(c[r2][f][i] + bv);
        }
    }
  }
}

// --------------------------- flash attention --------------------------------
// One block = 64 query rows of one (b,h). 128 thr = 4 waves; each wave owns a
// 16-row strip, 64-wide key tiles, online softmax, O accumulated in v8f.
// K tile stored [key][hd]; V tile stored transposed [hd][key] so both B-frags
// are contiguous.
__global__ __launch_bounds__(128)
void attn_kernel(const __bf16* __restrict__ Qb, const __bf16* __restrict__ Kb,
                 const __bf16* __restrict__ Vt, __bf16* __restrict__ Ob) {
  const int qt = blockIdx.x;
  const int h  = blockIdx.y;
  const int b  = blockIdx.z;
  const int tid = threadIdx.x;
  const int wv = tid >> 5, lane = tid & 31;
  const int m16 = lane & 15, hh = lane >> 4;
  const size_t headOff = ((size_t)(b * H_ + h)) * S_ * HD_;

  __shared__ __align__(16) __bf16 Qs[64][72];     // [qrow][hd]
  __shared__ __align__(16) __bf16 Ks[64][72];     // [key][hd]
  __shared__ __align__(16) __bf16 Vs[64][72];     // [hd][key]  (transposed)
  __shared__ __align__(16) __bf16 Ps[4][16][72];  // per-wave P strip

  const int r = tid >> 1, cb = (tid & 1) * 32;
  const __bf16* Krow = Kb + headOff + (size_t)r * HD_ + cb;   // + key*HD_
  const __bf16* Vrow = Vt + headOff + (size_t)r * S_ + cb;    // + key offset

  {
    const uint4* src = (const uint4*)(Qb + headOff + (size_t)(qt * 64 + r) * HD_ + cb);
    uint4* dst = (uint4*)&Qs[r][cb];
    dst[0] = src[0]; dst[1] = src[1]; dst[2] = src[2]; dst[3] = src[3];
  }
  __syncthreads();

  v16bf aq0 = load_A_frag(&Qs[0][0], 72, wv * 16 + m16, 0,  hh);
  v16bf aq1 = load_A_frag(&Qs[0][0], 72, wv * 16 + m16, 32, hh);

  v8f o[4] = {};
  float mrow[8], lrow[8];
#pragma unroll
  for (int i = 0; i < 8; ++i) { mrow[i] = -1e30f; lrow[i] = 0.0f; }

  for (int j = 0; j < S_; j += 64) {
    {
      const uint4* sk = (const uint4*)(Krow + (size_t)j * HD_);
      uint4* dk = (uint4*)&Ks[r][cb];
      dk[0] = sk[0]; dk[1] = sk[1]; dk[2] = sk[2]; dk[3] = sk[3];
      const uint4* sv = (const uint4*)(Vrow + j);
      uint4* dv = (uint4*)&Vs[r][cb];
      dv[0] = sv[0]; dv[1] = sv[1]; dv[2] = sv[2]; dv[3] = sv[3];
    }
    if (j + 64 < S_) {                        // prefetch next K/V tile
      __builtin_prefetch(Krow + (size_t)(j + 64) * HD_, 0, 0);
      __builtin_prefetch(Vrow + j + 64, 0, 0);
    }
    __syncthreads();

    // S = (Q @ K^T) * scale  -> 16x64 per wave (4 fragments)
    v8f sf[4];
#pragma unroll
    for (int f = 0; f < 4; ++f) {
      v8f acc = {};
      v16bf bk0 = load_Bt_frag(&Ks[0][0], 72, 0,  f * 16 + m16, hh);
      acc = WMMA_BF16(aq0, bk0, acc);
      v16bf bk1 = load_Bt_frag(&Ks[0][0], 72, 32, f * 16 + m16, hh);
      acc = WMMA_BF16(aq1, bk1, acc);
#pragma unroll
      for (int i = 0; i < 8; ++i) acc[i] *= SCALE_;
      sf[f] = acc;
    }

    // online softmax (rows live in VGPR index i + 8*hh; N across 16 lanes)
    float mnew[8], alpha[8], rs[8];
#pragma unroll
    for (int i = 0; i < 8; ++i) {
      float rm = fmaxf(fmaxf(sf[0][i], sf[1][i]), fmaxf(sf[2][i], sf[3][i]));
      rm = redmax16(rm);
      mnew[i]  = fmaxf(mrow[i], rm);
      alpha[i] = __expf(mrow[i] - mnew[i]);
      rs[i] = 0.0f;
    }
#pragma unroll
    for (int f = 0; f < 4; ++f) {
#pragma unroll
      for (int i = 0; i < 8; ++i) {
        float p = __expf(sf[f][i] - mnew[i]);
        rs[i] += p;
        Ps[wv][i + 8 * hh][f * 16 + m16] = (__bf16)p;
      }
    }
#pragma unroll
    for (int i = 0; i < 8; ++i) {
      lrow[i] = alpha[i] * lrow[i] + redsum16(rs[i]);
      mrow[i] = mnew[i];
    }
    __syncthreads();

    // O = alpha*O + P @ V
#pragma unroll
    for (int f = 0; f < 4; ++f)
#pragma unroll
      for (int i = 0; i < 8; ++i) o[f][i] *= alpha[i];

#pragma unroll
    for (int kk = 0; kk < 64; kk += 32) {
      v16bf ap = load_A_frag(&Ps[wv][0][0], 72, m16, kk, hh);
#pragma unroll
      for (int f = 0; f < 4; ++f) {
        // B = V[key][hd] = Vs[hd][key]: col = hd (f*16+m16), run over keys
        v16bf bvv = load_Bt_frag(&Vs[0][0], 72, kk, f * 16 + m16, hh);
        o[f] = WMMA_BF16(ap, bvv, o[f]);
      }
    }
    __syncthreads();
  }

  // write concat-head bf16 [B,S,D]
#pragma unroll
  for (int f = 0; f < 4; ++f) {
#pragma unroll
    for (int i = 0; i < 8; ++i) {
      int row = qt * 64 + wv * 16 + i + 8 * hh;
      int col = h * HD_ + f * 16 + m16;
      Ob[((size_t)(b * S_ + row)) * D_ + col] = (__bf16)(o[f][i] / lrow[i]);
    }
  }
}

// --------------------------- output projection ------------------------------
// out[8192,1024] = Ab @ Wpt^T + bias ; Wpt is [1024][1024] N-major.
__global__ __launch_bounds__(256)
void proj_gemm_kernel(const __bf16* __restrict__ X, const __bf16* __restrict__ Wt,
                      const float* __restrict__ bias, float* __restrict__ out) {
  const int nBase = blockIdx.x * 128;
  const int mBase = blockIdx.y * 128;
  const int tid  = threadIdx.x;
  const int wv   = tid >> 5, lane = tid & 31;
  const int m16  = lane & 15, hh = lane >> 4;
  const int wrow = (wv & 3) * 32;
  const int wcol = (wv >> 2) * 64;

  __shared__ __align__(16) __bf16 As[2][128][40];
  __shared__ __align__(16) __bf16 Bs[2][128][40];

  const int ar = tid >> 1, acb = (tid & 1) * 16;
  const __bf16* Xrow = X  + (size_t)(mBase + ar) * D_ + acb;
  const __bf16* Wrow = Wt + (size_t)(nBase + ar) * D_ + acb;

  uint4 xr0, xr1, wr0, wr1;
  { const uint4* s = (const uint4*)(Xrow); xr0 = s[0]; xr1 = s[1]; }
  { const uint4* s = (const uint4*)(Wrow); wr0 = s[0]; wr1 = s[1]; }
  { uint4* d = (uint4*)&As[0][ar][acb]; d[0] = xr0; d[1] = xr1; }
  { uint4* d = (uint4*)&Bs[0][ar][acb]; d[0] = wr0; d[1] = wr1; }

  v8f c[2][4] = {};
  int cur = 0;

  for (int k0 = 0; k0 < D_; k0 += 32) {
    __syncthreads();
    const bool hasNext = (k0 + 32) < D_;
    if (hasNext) {
      { const uint4* s = (const uint4*)(Xrow + k0 + 32); xr0 = s[0]; xr1 = s[1]; }
      { const uint4* s = (const uint4*)(Wrow + k0 + 32); wr0 = s[0]; wr1 = s[1]; }
      if (k0 + 64 < D_) {
        __builtin_prefetch(Xrow + k0 + 64, 0, 0);
        __builtin_prefetch(Wrow + k0 + 64, 0, 0);
      }
    }

    v16bf a0 = load_A_frag(&As[cur][0][0], 40, wrow + m16, 0, hh);
    v16bf a1 = load_A_frag(&As[cur][0][0], 40, wrow + 16 + m16, 0, hh);
#pragma unroll
    for (int f = 0; f < 4; ++f) {
      v16bf b = load_Bt_frag(&Bs[cur][0][0], 40, 0, wcol + f * 16 + m16, hh);
      c[0][f] = WMMA_BF16(a0, b, c[0][f]);
      c[1][f] = WMMA_BF16(a1, b, c[1][f]);
    }

    if (hasNext) {
      { uint4* d = (uint4*)&As[cur ^ 1][ar][acb]; d[0] = xr0; d[1] = xr1; }
      { uint4* d = (uint4*)&Bs[cur ^ 1][ar][acb]; d[0] = wr0; d[1] = wr1; }
      cur ^= 1;
    }
  }

#pragma unroll
  for (int f = 0; f < 4; ++f) {
    int cc = nBase + wcol + f * 16 + m16;
    float bv = bias[cc];
#pragma unroll
    for (int r2 = 0; r2 < 2; ++r2)
#pragma unroll
      for (int i = 0; i < 8; ++i) {
        int r = mBase + wrow + r2 * 16 + i + 8 * hh;
        out[(size_t)r * D_ + cc] = c[r2][f][i] + bv;
      }
  }
}

// ------------------------------- launcher -----------------------------------
extern "C" void kernel_launch(void* const* d_in, const int* in_sizes, int n_in,
                              void* d_out, int out_size, void* d_ws, size_t ws_size,
                              hipStream_t stream) {
  const float* x     = (const float*)d_in[0];   // [B,S,D]
  const float* Wqkv  = (const float*)d_in[1];   // [D,3D]
  const float* bqkv  = (const float*)d_in[2];   // [3D]
  const float* Wproj = (const float*)d_in[3];   // [D,D]
  const float* bproj = (const float*)d_in[4];   // [D]
  float* out = (float*)d_out;

  // workspace carve (bf16 elements), all offsets 256B-aligned
  char* ws = (char*)d_ws;
  const size_t nX  = (size_t)B_ * S_ * D_;      // 8388608
  const size_t nWq = (size_t)D_ * N3_;          // 3145728
  const size_t nWp = (size_t)D_ * D_;           // 1048576
  __bf16* xb  = (__bf16*)(ws);                               // [B*S][D]
  __bf16* wqt = (__bf16*)(ws + 2 * nX);                      // [3D][D] transposed
  __bf16* wpt = (__bf16*)(ws + 2 * (nX + nWq));              // [D][D] transposed
  __bf16* Qb  = (__bf16*)(ws + 2 * (nX + nWq + nWp));        // [B,H,S,HD]
  __bf16* Kb  = (__bf16*)(ws + 2 * (2 * nX + nWq + nWp));    // [B,H,S,HD]
  __bf16* Vt  = (__bf16*)(ws + 2 * (3 * nX + nWq + nWp));    // [B,H,HD,S]
  __bf16* Ab  = (__bf16*)(ws + 2 * (4 * nX + nWq + nWp));    // [B,S,D]

  // 1) fp32 -> bf16 conversions (weights transposed to N-major)
  f32_to_bf16_kernel<<<(int)((nX + 255) / 256), 256, 0, stream>>>(x, xb, (int)nX);
  f32_to_bf16_T_kernel<<<dim3(N3_ / 32, D_ / 32), 256, 0, stream>>>(Wqkv, wqt, D_, N3_);
  f32_to_bf16_T_kernel<<<dim3(D_ / 32, D_ / 32), 256, 0, stream>>>(Wproj, wpt, D_, D_);

  // 2) QKV projection -> Q/K [B,H,S,HD], V^T [B,H,HD,S] bf16
  qkv_gemm_kernel<<<dim3(N3_ / 128, (B_ * S_) / 128), 256, 0, stream>>>(
      xb, wqt, bqkv, Qb, Kb, Vt);

  // 3) flash attention -> Ab [B,S,D] bf16
  attn_kernel<<<dim3(S_ / 64, H_, B_), 128, 0, stream>>>(Qb, Kb, Vt, Ab);

  // 4) output projection -> out fp32
  proj_gemm_kernel<<<dim3(D_ / 128, (B_ * S_) / 128), 256, 0, stream>>>(
      Ab, wpt, bproj, out);
}
